// VectorQuantizer_61125974556916
// MI455X (gfx1250) — compile-verified, matrix operand order checked by reading, not run on hardware
//
#include <hip/hip_runtime.h>

// ---------------------------------------------------------------------------
// VQ-VAE vector quantizer for MI455X (gfx1250, wave32, WMMA + async-LDS).
//   z   : [B=16384, D=512] f32
//   emb : [K=8192,  D=512] f32
// out = concat( z_q_st [B*D] f32, indices [B] (as f32), vq_loss [1] f32 )
//
// scores = ||e_k||^2 - 2*(z.e_k); GEMM via bf16 hi/lo split
// (3x V_WMMA_F32_16X16X32_BF16: hi*hi + hi*lo + lo*hi ~ fp32 dot).
// emb is pre-converted ONCE to packed bf16 hi/lo planes (live in L2);
// the argmin hot loop streams them into double-buffered LDS with
// GLOBAL_LOAD_ASYNC_TO_LDS_B128 (ASYNCcnt) overlapped with WMMA.
// ---------------------------------------------------------------------------

#define B_ROWS   16384
#define KCODES   8192
#define DDIM     512
#define WG_ROWS  64                 // rows per workgroup (4 waves x 16)
#define CK       32                 // codes staged in LDS per chunk
#define NCHUNK   (KCODES / CK)      // 256
#define NT       (CK / 16)          // 2 code tiles of 16 per chunk
#define NS       (DDIM / 32)        // 16 K-steps of 32
#define ROW_U    (DDIM / 2)         // 256 uints (packed bf16 pairs) per code row
#define LDS_STRIDE 260              // uints per LDS code row (256 + 4 pad -> bank rotate)

#define AS1 __attribute__((address_space(1)))
#define AS3 __attribute__((address_space(3)))

#if defined(__HIP_DEVICE_COMPILE__) && \
    __has_builtin(__builtin_amdgcn_global_load_async_to_lds_b128) && \
    __has_builtin(__builtin_amdgcn_s_wait_asynccnt)
#define HAVE_ASYNC 1
#else
#define HAVE_ASYNC 0
#endif

typedef __attribute__((ext_vector_type(16))) __bf16 v16bf;
typedef __attribute__((ext_vector_type(8)))  float  v8f;
typedef __attribute__((ext_vector_type(4)))  int    v4i;

union BfFrag { unsigned u[8]; v16bf v; };

__device__ __forceinline__ unsigned bf16_rne(float f) {
    unsigned u = __float_as_uint(f);
    unsigned r = 0x7FFFu + ((u >> 16) & 1u);
    return (u + r) >> 16;
}
__device__ __forceinline__ float bf16_to_f32(unsigned h) {
    return __uint_as_float(h << 16);
}
__device__ __forceinline__ unsigned pack_hi(float a, float b) {
    return bf16_rne(a) | (bf16_rne(b) << 16);
}
__device__ __forceinline__ unsigned pack_lo(float a, float b) {
    float ra = a - bf16_to_f32(bf16_rne(a));
    float rb = b - bf16_to_f32(bf16_rne(b));
    return bf16_rne(ra) | (bf16_rne(rb) << 16);
}

__device__ __forceinline__ v8f wmma_bf16(const unsigned* a, const unsigned* b, v8f c) {
    BfFrag fa, fb;
#pragma unroll
    for (int i = 0; i < 8; ++i) { fa.u[i] = a[i]; fb.u[i] = b[i]; }
    return __builtin_amdgcn_wmma_f32_16x16x32_bf16(
        /*neg_a=*/false, fa.v, /*neg_b=*/false, fb.v,
        /*c_mod=*/(short)0, c, /*reuse_a=*/false, /*reuse_b=*/false);
}

// 16-byte global -> LDS async copy (ASYNCcnt) with sync fallback.
__device__ __forceinline__ void cp16_g2l(const void* g, void* l) {
#if HAVE_ASYNC
    __builtin_amdgcn_global_load_async_to_lds_b128((AS1 v4i*)g, (AS3 v4i*)l, 0, 0);
#else
    *(uint4*)l = *(const uint4*)g;
#endif
}
__device__ __forceinline__ void wait_async_all() {
#if HAVE_ASYNC
    __builtin_amdgcn_s_wait_asynccnt(0);
#endif
}

// ---------------------------------------------------------------------------
// Kernel P: convert emb (f32) -> packed bf16 hi/lo planes [K][256] uints each.
// One-time cost; planes (16 MB) stay resident in the 192 MB L2.
// ---------------------------------------------------------------------------
__global__ __launch_bounds__(256) void vq_prep_emb_kernel(
        const float* __restrict__ emb,
        unsigned* __restrict__ plane_hi, unsigned* __restrict__ plane_lo) {
    const size_t i = (size_t)blockIdx.x * 256 + threadIdx.x;  // one packed pair
    const float2 f = *(const float2*)(emb + 2 * i);
    plane_hi[i] = pack_hi(f.x, f.y);
    plane_lo[i] = pack_lo(f.x, f.y);
}

// ---------------------------------------------------------------------------
// Kernel 0: code norms ||e_k||^2  (one wave32 per code row)
// ---------------------------------------------------------------------------
__global__ __launch_bounds__(256) void vq_enorm_kernel(
        const float* __restrict__ emb, float* __restrict__ enorm) {
    const int code = blockIdx.x * 8 + (threadIdx.x >> 5);
    const int lane = threadIdx.x & 31;
    const float* e = emb + (size_t)code * DDIM;
    float s = 0.0f;
#pragma unroll
    for (int j = 0; j < 4; ++j) {
        float4 q = *(const float4*)(e + j * 128 + lane * 4);
        s += q.x * q.x + q.y * q.y + q.z * q.z + q.w * q.w;
    }
#pragma unroll
    for (int m = 16; m >= 1; m >>= 1) s += __shfl_xor(s, m, 32);
    if (lane == 0) enorm[code] = s;
}

// ---------------------------------------------------------------------------
// Stage one emb chunk (CK code rows, hi+lo planes) into LDS.
// 128 threads, 16B units; per-lane LDS addresses apply the padded stride.
// ---------------------------------------------------------------------------
__device__ __forceinline__ void stage_chunk(
        const unsigned* __restrict__ plane_hi, const unsigned* __restrict__ plane_lo,
        unsigned* dst_hi, unsigned* dst_lo, int c0, int tid) {
#pragma unroll
    for (int j = 0; j < (CK * (ROW_U / 4)) / 128; ++j) {   // 16 units / thread / plane
        const int u = j * 128 + tid;
        const int r = u >> 6;          // code row in chunk
        const int c = u & 63;          // 16B unit within row
        cp16_g2l(plane_hi + (size_t)(c0 + r) * ROW_U + c * 4,
                 dst_hi + r * LDS_STRIDE + c * 4);
        cp16_g2l(plane_lo + (size_t)(c0 + r) * ROW_U + c * 4,
                 dst_lo + r * LDS_STRIDE + c * 4);
    }
}

// ---------------------------------------------------------------------------
// Kernel 1: distances + argmin via WMMA, double-buffered async LDS staging.
//  - 4 waves / WG; wave owns 16 z rows; A-frags (bf16 hi/lo, all D) in VGPRs.
//  - chunk ch+1 streams into LDS buffer (ch+1)&1 while WMMA consumes ch&1.
// ---------------------------------------------------------------------------
__global__ __launch_bounds__(128) void vq_argmin_kernel(
        const float* __restrict__ z,
        const unsigned* __restrict__ plane_hi, const unsigned* __restrict__ plane_lo,
        const float* __restrict__ enorm,
        int* __restrict__ idx_out, float* __restrict__ idx_f_out) {

    __shared__ unsigned lds_hi[2][CK * LDS_STRIDE];
    __shared__ unsigned lds_lo[2][CK * LDS_STRIDE];

    const int tid  = threadIdx.x;
    const int wave = tid >> 5;
    const int lane = tid & 31;
    const int half = lane >> 4;     // K-half selector for A/B fragments
    const int nl   = lane & 15;     // row (A) / col (B,C) within tile
    const int wg_row0 = blockIdx.x * WG_ROWS;
    const int arow = wg_row0 + wave * 16 + nl;

    // ---- A fragments: 16 z rows, bf16 hi/lo, kept entirely in VGPRs -------
    // 16-bit A layout: lanes 0-15 hold M with K = {k0..k0+7, k0+16..k0+23},
    // lanes 16-31 the +8 shifted ranges (half*8).
    unsigned a_hi[NS][8], a_lo[NS][8];
    const float* zp = z + (size_t)arow * DDIM;
#pragma unroll
    for (int s = 0; s < NS; ++s) {
        const int k0 = s * 32 + half * 8;
#pragma unroll
        for (int c = 0; c < 2; ++c) {            // c=0 -> K=k0.., c=1 -> K=k0+16..
            float4 f0 = *(const float4*)(zp + k0 + c * 16);
            float4 f1 = *(const float4*)(zp + k0 + c * 16 + 4);
            a_hi[s][4 * c + 0] = pack_hi(f0.x, f0.y);
            a_hi[s][4 * c + 1] = pack_hi(f0.z, f0.w);
            a_hi[s][4 * c + 2] = pack_hi(f1.x, f1.y);
            a_hi[s][4 * c + 3] = pack_hi(f1.z, f1.w);
            a_lo[s][4 * c + 0] = pack_lo(f0.x, f0.y);
            a_lo[s][4 * c + 1] = pack_lo(f0.z, f0.w);
            a_lo[s][4 * c + 2] = pack_lo(f1.x, f1.y);
            a_lo[s][4 * c + 3] = pack_lo(f1.z, f1.w);
        }
    }

    // Running (min, argmin): C layout -> lane (half,nl) + VGPR v covers
    // row M = half*8+v, col N = nl; codes visited in increasing order so
    // strict '<' keeps the first minimum per lane column.
    float bestv[8];
    int   besti[8];
#pragma unroll
    for (int v = 0; v < 8; ++v) { bestv[v] = 3.4e38f; besti[v] = 0; }

    // prologue: stage chunk 0
    stage_chunk(plane_hi, plane_lo, lds_hi[0], lds_lo[0], 0, tid);
    wait_async_all();
    __syncthreads();

    for (int ch = 0; ch < NCHUNK; ++ch) {
        const int c0  = ch * CK;
        const int cur = ch & 1;

        // kick off async staging of the next chunk into the other buffer
        if (ch + 1 < NCHUNK)
            stage_chunk(plane_hi, plane_lo, lds_hi[cur ^ 1], lds_lo[cur ^ 1],
                        c0 + CK, tid);

        // ---- WMMA: acc[nt] = z_tile . emb_tile^T (bf16 3-term split) -------
        v8f acc[NT] = {};
#pragma unroll
        for (int s = 0; s < NS; ++s) {
#pragma unroll
            for (int nt = 0; nt < NT; ++nt) {
                const unsigned* bh =
                    &lds_hi[cur][(nt * 16 + nl) * LDS_STRIDE + s * 16 + half * 4];
                const unsigned* bl =
                    &lds_lo[cur][(nt * 16 + nl) * LDS_STRIDE + s * 16 + half * 4];
                uint4 h0 = *(const uint4*)(bh);
                uint4 h1 = *(const uint4*)(bh + 8);
                uint4 l0 = *(const uint4*)(bl);
                uint4 l1 = *(const uint4*)(bl + 8);
                unsigned bhi[8] = {h0.x, h0.y, h0.z, h0.w, h1.x, h1.y, h1.z, h1.w};
                unsigned blo[8] = {l0.x, l0.y, l0.z, l0.w, l1.x, l1.y, l1.z, l1.w};
                acc[nt] = wmma_bf16(a_hi[s], bhi, acc[nt]);   // hi*hi
                acc[nt] = wmma_bf16(a_hi[s], blo, acc[nt]);   // hi*lo
                acc[nt] = wmma_bf16(a_lo[s], bhi, acc[nt]);   // lo*hi
            }
        }

        // ---- fold chunk scores into running argmin -------------------------
#pragma unroll
        for (int nt = 0; nt < NT; ++nt) {
            const int code = c0 + nt * 16 + nl;
            const float en = enorm[code];
#pragma unroll
            for (int v = 0; v < 8; ++v) {
                float sc = en - 2.0f * acc[nt][v];
                if (sc < bestv[v]) { bestv[v] = sc; besti[v] = code; }
            }
        }

        wait_async_all();   // next buffer fully landed (this wave's copies)
        __syncthreads();    // ...and everyone else's; also fences buffer reuse
    }

    // ---- cross-lane reduction over the 16 N-columns (stay inside halves) --
#pragma unroll
    for (int v = 0; v < 8; ++v) {
        float bv = bestv[v];
        int   bi = besti[v];
#pragma unroll
        for (int m = 1; m <= 8; m <<= 1) {
            float ov = __shfl_xor(bv, m, 32);
            int   oi = __shfl_xor(bi, m, 32);
            if (ov < bv || (ov == bv && oi < bi)) { bv = ov; bi = oi; }
        }
        if (nl == 0) {
            int r = wg_row0 + wave * 16 + half * 8 + v;
            idx_out[r]   = bi;
            idx_f_out[r] = (float)bi;
        }
    }
}

// ---------------------------------------------------------------------------
// Kernel 2: gather z_q = emb[idx], write z_q_st (== z_q numerically) and
//           per-row partial sums of (z_q - z)^2 (deterministic tree).
// ---------------------------------------------------------------------------
__global__ __launch_bounds__(256) void vq_gather_kernel(
        const float* __restrict__ z, const float* __restrict__ emb,
        const int* __restrict__ idx, float* __restrict__ zq_out,
        float* __restrict__ partials) {
    const int row = blockIdx.x;
    const int t   = threadIdx.x;
    const int code = idx[row];
    const float* e  = emb + (size_t)code * DDIM;
    const float* zr = z   + (size_t)row  * DDIM;
    float s = 0.0f;
#pragma unroll
    for (int j = 0; j < 2; ++j) {
        int d = t + j * 256;
        float ev = e[d];
        float df = ev - zr[d];
        zq_out[(size_t)row * DDIM + d] = ev;
        s += df * df;
    }
#pragma unroll
    for (int m = 16; m >= 1; m >>= 1) s += __shfl_xor(s, m, 32);
    __shared__ float red[8];
    if ((t & 31) == 0) red[t >> 5] = s;
    __syncthreads();
    if (t == 0) {
        float tot = 0.0f;
#pragma unroll
        for (int w = 0; w < 8; ++w) tot += red[w];
        partials[row] = tot;
    }
}

// ---------------------------------------------------------------------------
// Kernel 3: vq_loss = 1.25 * sum(partials) / (B*D)   (single block, fixed order)
// ---------------------------------------------------------------------------
__global__ __launch_bounds__(256) void vq_finalize_kernel(
        const float* __restrict__ partials, float* __restrict__ loss_out) {
    __shared__ float red[256];
    const int t = threadIdx.x;
    float s = 0.0f;
    for (int j = 0; j < B_ROWS / 256; ++j) s += partials[t + j * 256];
    red[t] = s;
    __syncthreads();
    for (int off = 128; off >= 1; off >>= 1) {
        if (t < off) red[t] += red[t + off];
        __syncthreads();
    }
    if (t == 0) loss_out[0] = 1.25f * red[0] / (float)((size_t)B_ROWS * DDIM);
}

// ---------------------------------------------------------------------------
extern "C" void kernel_launch(void* const* d_in, const int* in_sizes, int n_in,
                              void* d_out, int out_size, void* d_ws, size_t ws_size,
                              hipStream_t stream) {
    const float* z   = (const float*)d_in[0];   // [16384*512]
    const float* emb = (const float*)d_in[1];   // [8192*512]

    float* out   = (float*)d_out;
    float* zq    = out;                                   // [B*D]
    float* idxf  = out + (size_t)B_ROWS * DDIM;           // [B]
    float* loss  = idxf + B_ROWS;                         // [1]

    int*      ws_idx   = (int*)d_ws;                              // [B]
    float*    ws_enorm = (float*)d_ws + B_ROWS;                   // [K]
    float*    ws_part  = (float*)d_ws + B_ROWS + KCODES;          // [B]
    unsigned* plane_hi = (unsigned*)d_ws + B_ROWS + KCODES + B_ROWS;  // [K*256]
    unsigned* plane_lo = plane_hi + (size_t)KCODES * ROW_U;           // [K*256]

    vq_prep_emb_kernel<<<(KCODES * ROW_U) / 256, 256, 0, stream>>>(emb, plane_hi,
                                                                   plane_lo);
    vq_enorm_kernel<<<KCODES / 8, 256, 0, stream>>>(emb, ws_enorm);
    vq_argmin_kernel<<<B_ROWS / WG_ROWS, 128, 0, stream>>>(z, plane_hi, plane_lo,
                                                           ws_enorm, ws_idx, idxf);
    vq_gather_kernel<<<B_ROWS, 256, 0, stream>>>(z, emb, ws_idx, zq, ws_part);
    vq_finalize_kernel<<<1, 256, 0, stream>>>(ws_part, loss);
}
